// SynthesisLayer_61564061221471
// MI455X (gfx1250) — matrix operand order repaired and checked by reading, not compiled, not run on hardware
//
#include <hip/hip_runtime.h>
#include <cstdint>

typedef __attribute__((ext_vector_type(16))) __bf16 v16bf;
typedef __attribute__((ext_vector_type(8)))  float  v8f;

#define B_    16
#define IC_   512
#define OC_   512
#define WDIM_ 512
#define RES_  64
#define NPOS  4096
#define PADW  66

// LDS B-slab: 128 positions x 128 ic (bf16), row stride 136 halfs (272B):
// 16B-aligned and bank-conflict-free for b128 fragment reads (68 dwords % 64 = 4).
#define BROW  136

union FragBF { uint4 q[2]; v16bf v; };

#if defined(__HIP_DEVICE_COMPILE__) && __has_builtin(__builtin_amdgcn_global_load_async_to_lds_b128)
#define ASYNC_COPY 1
#else
#define ASYNC_COPY 0
#endif

__device__ __forceinline__ void cp16_g2l(const unsigned short* g, unsigned short* l) {
#if ASYNC_COPY
  // Builtin params (from hipcc diagnostic): non-const int __vector(4) pointers,
  // global (AS1) source, LDS (AS3) dest, then imm offset + cpol.
  typedef int v4i_t __attribute__((__vector_size__(16)));
  typedef __attribute__((address_space(1))) v4i_t* gp1_t;
  typedef __attribute__((address_space(3))) v4i_t* lp3_t;
  // generic->AS3: low 32 bits of a generic LDS pointer are the LDS offset
  __builtin_amdgcn_global_load_async_to_lds_b128(
      (gp1_t)(uintptr_t)g, (lp3_t)(unsigned)(uintptr_t)l, 0, 0);
#else
  *(uint4*)l = *(const uint4*)g;
#endif
}

__device__ __forceinline__ void wait_async_copies() {
#if ASYNC_COPY
#if __has_builtin(__builtin_amdgcn_s_wait_asynccnt)
  __builtin_amdgcn_s_wait_asynccnt(0);
#else
  asm volatile("s_wait_asynccnt 0" ::: "memory");
#endif
#endif
}

__device__ __forceinline__ unsigned short f2bf(float f) {
  union { float f; unsigned u; } v; v.f = f;
  unsigned u = v.u;
  unsigned r = u + 0x7FFFu + ((u >> 16) & 1u);   // round-to-nearest-even
  return (unsigned short)(r >> 16);
}

// styles[b][ic] = w[b,:] . affine_weight[ic,:] / sqrt(512) + affine_bias[ic]
__global__ void k_styles(const float* __restrict__ w, const float* __restrict__ aw,
                         const float* __restrict__ ab, float* __restrict__ styles) {
  int idx = blockIdx.x * blockDim.x + threadIdx.x;
  if (idx >= B_ * IC_) return;
  int b = idx >> 9, ic = idx & 511;
  const float g = 0.04419417382415922f;  // 1/sqrt(512)
  const float* wr = w + (size_t)b * WDIM_;
  const float* ar = aw + (size_t)ic * WDIM_;
  float acc = 0.f;
  for (int d = 0; d < WDIM_; ++d) acc += wr[d] * ar[d];
  styles[idx] = acc * g + ab[ic];
}

// Pack conv_weight into per-lane WMMA-A fragment order (bf16) and compute S2[oc][ic]
__global__ void k_pack_w(const float* __restrict__ cw, float* __restrict__ S2,
                         unsigned short* __restrict__ wp) {
  int idx = blockIdx.x * blockDim.x + threadIdx.x;
  if (idx >= OC_ * IC_) return;
  int oc = idx >> 9, ic = idx & 511;
  int mTile = oc >> 4, m = oc & 15;
  int icBlk = ic >> 5, kk = ic & 31;
  int group = kk >> 3, lo = kk & 7;
  int lane = m + ((group & 1) << 4);     // K 8..15 / 24..31 live in lanes 16..31
  int half = lo + ((group >> 1) << 3);   // K 16..31 live in VGPRs 4..7
  const float* src = cw + (size_t)idx * 9;
  float ssq = 0.f;
  for (int t = 0; t < 9; ++t) {
    float v = src[t];
    ssq += v * v;
    size_t o = ((((size_t)t * 32 + mTile) * 16 + icBlk) * 32 + lane) * 16 + half;
    wp[o] = f2bf(v);
  }
  S2[idx] = ssq;
}

// dcoef[b][oc] = rsqrt(sum_ic S2[oc][ic]*styles[b][ic]^2 + 1e-8)
__global__ void k_dcoef(const float* __restrict__ S2, const float* __restrict__ styles,
                        float* __restrict__ dc) {
  int idx = blockIdx.x * blockDim.x + threadIdx.x;
  if (idx >= B_ * OC_) return;
  int b = idx >> 9, oc = idx & 511;
  const float* s2r = S2 + (size_t)oc * IC_;
  const float* str = styles + (size_t)b * IC_;
  float acc = 1e-8f;
  for (int ic = 0; ic < IC_; ++ic) { float s = str[ic]; acc += s2r[ic] * s * s; }
  dc[idx] = rsqrtf(acc);
}

__global__ void k_zero(uint4* __restrict__ p, int n) {
  int i = blockIdx.x * blockDim.x + threadIdx.x;
  if (i < n) p[i] = make_uint4(0u, 0u, 0u, 0u);
}

// xs[b][h+1][w+1][ic] = bf16(x[b][ic][h][w] * styles[b][ic])  (LDS transpose per 32-ic tile)
__global__ void k_modx(const float* __restrict__ x, const float* __restrict__ styles,
                       unsigned short* __restrict__ xs) {
  int h = blockIdx.x;            // 0..63
  int ic0 = blockIdx.y << 5;     // 0,32,..,480
  int b = blockIdx.z;
  __shared__ unsigned short lds[32][65];
  const float* xb = x + ((size_t)b * IC_ + ic0) * NPOS + (size_t)h * RES_;
  const float* st = styles + (size_t)b * IC_ + ic0;
  for (int e = threadIdx.x; e < 32 * RES_; e += 256) {
    int i = e >> 6, w = e & 63;
    lds[i][w] = f2bf(xb[(size_t)i * NPOS + w] * st[i]);
  }
  __syncthreads();
  unsigned short* ob = xs + ((size_t)b * PADW * PADW + (size_t)(h + 1) * PADW + 1) * IC_ + ic0;
  for (int e = threadIdx.x; e < 32 * RES_; e += 256) {
    int w = e >> 5, i = e & 31;
    ob[(size_t)w * IC_ + i] = lds[i][w];
  }
}

// Implicit-GEMM modulated conv: per sample, M=OC (tile 128), N=4096 (tile 128), K=9*512.
// 8 waves, wave w owns M rows [mBase+16w, +16), all 128 N.
// B staged per (tap, 128-ic slab) via async global->LDS; 32 WMMAs per barrier phase.
__global__ __launch_bounds__(256) void k_conv(
    const unsigned short* __restrict__ xs, const unsigned short* __restrict__ wp,
    const float* __restrict__ dc, const float* __restrict__ bias,
    const float* __restrict__ noise, const float* __restrict__ nsp,
    float* __restrict__ out) {
  const int b = blockIdx.z;
  const int mBase = blockIdx.x << 7;
  const int nBase = blockIdx.y << 7;
  const int tid = threadIdx.x;
  const int wave = tid >> 5;
  const int lane = tid & 31;

  __shared__ __align__(16) unsigned short Bs[128 * BROW];  // ~34 KB

  v8f acc[8] = {};

  const int mTile = (mBase >> 4) + wave;
  const size_t xbase = (size_t)b * (PADW * PADW * IC_);

  // staging geometry: 128 rows x 16 chunks of 16B; q fixed per thread, row steps by 16
  const int q = tid & 15;          // 16B chunk within 256B row
  const int row0 = tid >> 4;       // 0..15

  for (int t = 0; t < 9; ++t) {
    const int ky = t / 3, kx = t % 3;   // padded layout stored at +1
    for (int kg = 0; kg < 4; ++kg) {    // 128-ic slab = 4 kb blocks
      // --- stage B slab: 128 pos x 128 ic bf16 = 32KB ---
      #pragma unroll
      for (int it = 0; it < 8; ++it) {
        int n = row0 + (it << 4);
        int pos = nBase + n;
        int row = (pos >> 6) + ky;
        int col = (pos & 63) + kx;
        const unsigned short* src =
            xs + xbase + ((size_t)row * PADW + col) * IC_ + (kg << 7) + (q << 3);
        cp16_g2l(src, &Bs[n * BROW + (q << 3)]);
      }
      wait_async_copies();
      __syncthreads();

      #pragma unroll
      for (int kb2 = 0; kb2 < 4; ++kb2) {
        // A fragment: direct from packed global (weights are L2-resident, 4.5MB)
        FragBF af;
        const unsigned short* ap =
            wp + (((size_t)t * 32 + mTile) * 16 + ((kg << 2) + kb2)) * 512 + lane * 16;
        af.q[0] = *(const uint4*)(ap);
        af.q[1] = *(const uint4*)(ap + 8);

        #pragma unroll
        for (int j = 0; j < 8; ++j) {
          FragBF bfrag;
          int off = ((j << 4) + (lane & 15)) * BROW + (kb2 << 5) + ((lane >> 4) << 3);
          bfrag.q[0] = *(const uint4*)(&Bs[off]);
          bfrag.q[1] = *(const uint4*)(&Bs[off + 16]);
          acc[j] = __builtin_amdgcn_wmma_f32_16x16x32_bf16(
              false, af.v, false, bfrag.v, (short)0, acc[j], false, false);
        }
      }
      __syncthreads();
    }
  }

  // --- epilogue: demod * acc + noise + bias, lrelu(0.2)*sqrt(2) ---
  const float ns = nsp[0];
  const float gain = 1.4142135623730951f;
  float dcv[8], bv[8];
  int ocBase = mBase + (wave << 4) + ((lane >> 4) << 3);
  #pragma unroll
  for (int r = 0; r < 8; ++r) {
    dcv[r] = dc[(b << 9) + ocBase + r];
    bv[r] = bias[ocBase + r];
  }
  #pragma unroll
  for (int j = 0; j < 8; ++j) {
    int n = nBase + (j << 4) + (lane & 15);
    float nz = noise[n] * ns;
    #pragma unroll
    for (int r = 0; r < 8; ++r) {
      float y = acc[j][r] * dcv[r] + nz + bv[r];
      y = (y >= 0.f ? y : 0.2f * y) * gain;
      out[(((size_t)b << 9) + ocBase + r) * NPOS + n] = y;
    }
  }
}

extern "C" void kernel_launch(void* const* d_in, const int* in_sizes, int n_in,
                              void* d_out, int out_size, void* d_ws, size_t ws_size,
                              hipStream_t stream) {
  const float* x   = (const float*)d_in[0];
  const float* w   = (const float*)d_in[1];
  const float* cw  = (const float*)d_in[2];
  const float* aw  = (const float*)d_in[3];
  const float* ab  = (const float*)d_in[4];
  const float* bs  = (const float*)d_in[5];
  const float* nc  = (const float*)d_in[6];
  const float* nst = (const float*)d_in[7];
  float* out = (float*)d_out;

  char* base = (char*)d_ws;
  size_t cur = 0;
  auto alloc = [&](size_t bytes) -> char* {
    char* p = base + cur;
    cur += (bytes + 255) & ~(size_t)255;
    return p;
  };
  float* styles = (float*)alloc((size_t)B_ * IC_ * 4);
  float* dcoef  = (float*)alloc((size_t)B_ * OC_ * 4);
  float* S2     = (float*)alloc((size_t)OC_ * IC_ * 4);
  unsigned short* wp = (unsigned short*)alloc((size_t)9 * 32 * 16 * 32 * 16 * 2);   // 4.5 MB
  unsigned short* xs = (unsigned short*)alloc((size_t)B_ * PADW * PADW * IC_ * 2);  // 71.4 MB
  if (cur > ws_size) return;  // workspace too small: do nothing rather than corrupt memory

  k_styles<<<(B_ * IC_ + 255) / 256, 256, 0, stream>>>(w, aw, ab, styles);
  k_pack_w<<<(OC_ * IC_ + 255) / 256, 256, 0, stream>>>(cw, S2, wp);
  k_dcoef<<<(B_ * OC_ + 255) / 256, 256, 0, stream>>>(S2, styles, dcoef);

  int zeroN = (int)(((size_t)B_ * PADW * PADW * IC_ * 2) / 16);
  k_zero<<<(zeroN + 255) / 256, 256, 0, stream>>>((uint4*)xs, zeroN);

  dim3 gm(RES_, IC_ / 32, B_);
  k_modx<<<gm, 256, 0, stream>>>(x, styles, xs);

  dim3 gc(OC_ / 128, NPOS / 128, B_);
  k_conv<<<gc, 256, 0, stream>>>(xs, wp, dcoef, bs, nc, nst, out);
}
